// TransformerBlock_63763084477194
// MI455X (gfx1250) — matrix operand order repaired and checked by reading, not compiled, not run on hardware
//
#include <hip/hip_runtime.h>
#include <hip/hip_bf16.h>
#include <math.h>

// ---------------------------------------------------------------------------
// Problem dims (fixed by the reference)
// ---------------------------------------------------------------------------
#define B_   2
#define S_   2048
#define D_   1024
#define H_   16
#define DK_  64
#define DFF_ 4096
#define ROWS (B_ * S_)   // 4096 token rows

typedef __attribute__((ext_vector_type(16))) _Float16 v16h;
typedef __attribute__((ext_vector_type(8)))  _Float16 v8h;
typedef __attribute__((ext_vector_type(8)))  float    v8f;
typedef __attribute__((ext_vector_type(4)))  unsigned u32x4;
typedef __attribute__((ext_vector_type(8)))  int      i32x8;
typedef __attribute__((ext_vector_type(4)))  int      i32x4;

// ---------------------------------------------------------------------------
// WMMA helper (gfx1250: V_WMMA_F32_16X16X32_F16, wave32)
// ---------------------------------------------------------------------------
__device__ __forceinline__ v8f wmma_16x16x32(v16h a, v16h b, v8f c) {
    return __builtin_amdgcn_wmma_f32_16x16x32_f16(
        /*neg_a=*/false, a, /*neg_b=*/false, b,
        /*c_mod=*/(short)0, c, /*reuse_a=*/false, /*reuse_b=*/false);
}

// A fragment (16x32 f16, MxK). Lanes 0-15: M=lane, K={0..7,16..23};
// lanes 16-31: M=lane-16, K={8..15,24..31}. rowp = this lane's row.
__device__ __forceinline__ v16h frag_a(const _Float16* rowp, int lane) {
    const int o0 = (lane & 16) ? 8 : 0;
    v8h lo = *(const v8h*)(rowp + o0);
    v8h hi = *(const v8h*)(rowp + o0 + 16);
    v16h f;
#pragma unroll
    for (int i = 0; i < 8; ++i) { f[i] = lo[i]; f[i + 8] = hi[i]; }
    return f;
}

// B fragment (32x16 f16, KxN) from an N-major buffer: colp = this lane's
// column (N=lane&15), contiguous in K. Lanes 0-15: K={0..7,16..23}; 16-31: K={8..15,24..31}.
__device__ __forceinline__ v16h frag_b(const _Float16* colp, int lane) {
    const int o = (lane & 16) ? 8 : 0;
    v8h lo = *(const v8h*)(colp + o);
    v8h hi = *(const v8h*)(colp + o + 16);
    v16h f;
#pragma unroll
    for (int i = 0; i < 8; ++i) { f[i] = lo[i]; f[i + 8] = hi[i]; }
    return f;
}

// ---------------------------------------------------------------------------
// CDNA5 async memory->LDS copy (GLOBAL_LOAD_ASYNC_TO_LDS_B128, ASYNCcnt).
// LDS byte offset = low 32 bits of the flat shared-pointer (LDS aperture
// occupies ADDR[63:32] only, per the ISA aperture check).
// ---------------------------------------------------------------------------
__device__ __forceinline__ void async_copy_b128(const void* lds_dst,
                                                const void* global_src) {
    unsigned lds_off = (unsigned)(unsigned long long)lds_dst;
    unsigned long long gaddr = (unsigned long long)global_src;
    asm volatile("global_load_async_to_lds_b128 %0, %1, off"
                 :: "v"(lds_off), "v"(gaddr)
                 : "memory");
}
__device__ __forceinline__ void wait_asynccnt_0() {
    asm volatile("s_wait_asynccnt 0x0" ::: "memory");
}

// ---------------------------------------------------------------------------
// CDNA5 Tensor Data Mover: 2D f16 tile load, global -> LDS.
// Descriptor bitfields per ISA 8.3/8.4: G0 = {count=1 | lds_addr |
// global_addr[56:0] | type=2}; G1 = {mask=0, data_size=1(2B),
// tensor_dim0/1, tile_dim0/1, tensor_dim0_stride}.
// This toolchain's builtin takes 6 args (g0, g1, g2, g3, g_extra, cpol).
// ---------------------------------------------------------------------------
__device__ __forceinline__ void tdm_load_tile_f16(
    const void* lds_dst, const void* gsrc,
    unsigned tensor_d0, unsigned tensor_d1,
    unsigned tile_d0, unsigned tile_d1, unsigned row_stride_elems) {
    u32x4 g0;
    g0[0] = 1u;                                     // count=1, user mode
    g0[1] = (unsigned)(unsigned long long)lds_dst;  // lds_addr (byte offset)
    unsigned long long ga = (unsigned long long)gsrc;
    g0[2] = (unsigned)ga;                           // global_addr[31:0]
    g0[3] = ((unsigned)(ga >> 32) & 0x01FFFFFFu) | (2u << 30);  // [56:32]|type=2

    i32x8 g1;
    g1[0] = 1 << 16;                                // data_size=1 -> 2 bytes
    g1[1] = (int)((tensor_d0 & 0xFFFFu) << 16);     // tensor_dim0[15:0]
    g1[2] = (int)(((tensor_d0 >> 16) & 0xFFFFu) |
                  ((tensor_d1 & 0xFFFFu) << 16));   // dim0[31:16] | dim1[15:0]
    g1[3] = (int)(((tensor_d1 >> 16) & 0xFFFFu) |
                  ((tile_d0 & 0xFFFFu) << 16));     // dim1[31:16] | tile_dim0
    g1[4] = (int)(tile_d1 & 0xFFFFu);               // tile_dim1; tile_dim2=0
    g1[5] = (int)row_stride_elems;                  // tensor_dim0_stride[31:0]
    g1[6] = 0;                                      // stride[47:32]|d1stride lo
    g1[7] = 0;
    i32x4 gz4 = {0, 0, 0, 0};
    i32x8 gz8 = {0, 0, 0, 0, 0, 0, 0, 0};
#if __has_builtin(__builtin_amdgcn_tensor_load_to_lds)
    __builtin_amdgcn_tensor_load_to_lds(g0, g1, gz4, gz4, gz8, 0);
#else
    asm volatile("tensor_load_to_lds %0, %1" :: "s"(g0), "s"(g1) : "memory");
#endif
}
__device__ __forceinline__ void wait_tensorcnt_0() {
#if __has_builtin(__builtin_amdgcn_s_wait_tensorcnt)
    __builtin_amdgcn_s_wait_tensorcnt(0);
#else
    asm volatile("s_wait_tensorcnt 0x0" ::: "memory");
#endif
}

// ---------------------------------------------------------------------------
// fp32 -> f16 conversion with transpose: src is [K][N] row-major,
// dst is [N][K] row-major (N-major weights for the WMMA B path).
// ---------------------------------------------------------------------------
__global__ void cvt_transpose_f16(const float* __restrict__ src,
                                  _Float16* __restrict__ dst, int K, int N) {
    int i = blockIdx.x * blockDim.x + threadIdx.x;
    if (i < K * N) {
        int k = i / N, n = i % N;
        dst[(size_t)n * K + k] = (_Float16)src[i];
    }
}

// ---------------------------------------------------------------------------
// RMSNorm: one block per token row, writes f16 normalized activations
// ---------------------------------------------------------------------------
__global__ __launch_bounds__(256) void rmsnorm_f16(const float* __restrict__ x,
                                                   const float* __restrict__ g,
                                                   _Float16* __restrict__ out) {
    const int row = blockIdx.x;
    const int t   = threadIdx.x;
    const float* xr = x + (size_t)row * D_;
    float v0[4];
    float ss = 0.f;
#pragma unroll
    for (int j = 0; j < 4; ++j) {
        float v = xr[t + j * 256];
        v0[j] = v;
        ss += v * v;
    }
    __shared__ float red[256];
    red[t] = ss;
    __syncthreads();
    for (int off = 128; off > 0; off >>= 1) {
        if (t < off) red[t] += red[t + off];
        __syncthreads();
    }
    const float scale = rsqrtf(red[0] / (float)D_ + 1e-6f);
    _Float16* orow = out + (size_t)row * D_;
#pragma unroll
    for (int j = 0; j < 4; ++j)
        orow[t + j * 256] = (_Float16)(v0[j] * scale * g[t + j * 256]);
}

// ---------------------------------------------------------------------------
// Tiled WMMA GEMM: C[M,N] = A[M,K] * BT[N,K]^T, f16 in, f32 accumulate.
// Block tile 128x128, K-step 32. Tile staging via the Tensor Data Mover
// (one tensor_load_to_lds per matrix per K-step, issued by wave 0),
// double-buffered so the DMA overlaps the 8 WMMAs of the current tile.
// MODE 0: store f16.          MODE 1: f32 out = acc + resid.
// MODE 2: silu(acc)*aux->f16. MODE 3: scatter V^T per (b,h): [b,h,d,s] f16.
// ---------------------------------------------------------------------------
template <int MODE>
__global__ __launch_bounds__(256) void gemm_wmma(
    const _Float16* __restrict__ A, const _Float16* __restrict__ BT,
    _Float16* __restrict__ outh, float* __restrict__ outf,
    const float* __restrict__ resid, const _Float16* __restrict__ aux,
    int M, int N, int K) {
    const int BK = 32;
    __shared__ __align__(16) _Float16 As[2][128][BK];
    __shared__ __align__(16) _Float16 Bs[2][128][BK];

    const int tid  = threadIdx.x;
    const int lane = tid & 31;
    const int wave = tid >> 5;
    const int m0 = blockIdx.y * 128;
    const int n0 = blockIdx.x * 128;
    const int wm = (wave & 3) * 32;   // 4 waves along M
    const int wn = (wave >> 2) * 64;  // 2 waves along N

    const _Float16* Abase = A + (size_t)m0 * K;
    const _Float16* Bbase = BT + (size_t)n0 * K;

    const v8f vzero = {0.f, 0.f, 0.f, 0.f, 0.f, 0.f, 0.f, 0.f};
    v8f acc[2][4];
#pragma unroll
    for (int mt = 0; mt < 2; ++mt)
#pragma unroll
        for (int nt = 0; nt < 4; ++nt) acc[mt][nt] = vzero;

    const int nk = K / BK;
    if (wave == 0) {
        tdm_load_tile_f16(&As[0][0][0], Abase, K, M, BK, 128, K);
        tdm_load_tile_f16(&Bs[0][0][0], Bbase, K, N, BK, 128, K);
        wait_tensorcnt_0();
    }
    __syncthreads();

    for (int t = 0; t < nk; ++t) {
        const int buf = t & 1;
        if (wave == 0 && t + 1 < nk) {
            // Kick the DMA for the next K-panel; overlaps with compute below.
            tdm_load_tile_f16(&As[buf ^ 1][0][0], Abase + (t + 1) * BK,
                              K, M, BK, 128, K);
            tdm_load_tile_f16(&Bs[buf ^ 1][0][0], Bbase + (t + 1) * BK,
                              K, N, BK, 128, K);
            // Speculative prefetch of the panel after next (CDNA5 prefetch)
            __builtin_prefetch(Abase + (size_t)(t + 2) * BK, 0, 0);
            __builtin_prefetch(Bbase + (size_t)(t + 2) * BK, 0, 0);
        }

        v16h af[2], bf[4];
#pragma unroll
        for (int mt = 0; mt < 2; ++mt)
            af[mt] = frag_a(&As[buf][wm + mt * 16 + (lane & 15)][0], lane);
#pragma unroll
        for (int nt = 0; nt < 4; ++nt)
            bf[nt] = frag_b(&Bs[buf][wn + nt * 16 + (lane & 15)][0], lane);
#pragma unroll
        for (int mt = 0; mt < 2; ++mt)
#pragma unroll
            for (int nt = 0; nt < 4; ++nt)
                acc[mt][nt] = wmma_16x16x32(af[mt], bf[nt], acc[mt][nt]);

        if (wave == 0) wait_tensorcnt_0();   // next panel landed in LDS
        __syncthreads();
    }

    // Epilogue. C layout: lane holds column N=lane&15; VGPR r holds row
    // r (lanes 0-15) or r+8 (lanes 16-31).
    const int col = lane & 15;
    const int rb  = (lane & 16) ? 8 : 0;
#pragma unroll
    for (int mt = 0; mt < 2; ++mt)
#pragma unroll
        for (int nt = 0; nt < 4; ++nt) {
            const int gn = n0 + wn + nt * 16 + col;
#pragma unroll
            for (int r = 0; r < 8; ++r) {
                const int gm = m0 + wm + mt * 16 + rb + r;
                const size_t idx = (size_t)gm * N + gn;
                const float vv = acc[mt][nt][r];
                if (MODE == 0) {
                    outh[idx] = (_Float16)vv;
                } else if (MODE == 1) {
                    outf[idx] = vv + resid[idx];
                } else if (MODE == 2) {
                    const float u = (float)aux[idx];
                    const float sg = 1.f / (1.f + __expf(-vv));
                    outh[idx] = (_Float16)(vv * sg * u);
                } else {
                    // V^T scatter: gm = b*S + s ; gn = h*64 + d
                    const int bb = gm >> 11, si = gm & (S_ - 1);
                    const int hh = gn >> 6,  dd = gn & 63;
                    outh[(((size_t)bb * H_ + hh) * 64 + dd) * S_ + si] =
                        (_Float16)vv;
                }
            }
        }
}

// ---------------------------------------------------------------------------
// Attention: one wave (32 threads) per (b, h, 16-query tile).
// Strict causal mask (k < q, matching triu(diag=0) in the reference),
// softmax = exp(s) / (sum exp + 1e-9), no max subtraction (faithful).
// Q,K stored [b*S+s][H*DK]; V pre-transposed as vT[b][h][d][s] (f16).
// K chunks are staged into LDS with async global->LDS copies (ASYNCcnt).
// ---------------------------------------------------------------------------
__global__ __launch_bounds__(32) void attn_wmma(const _Float16* __restrict__ q,
                                                const _Float16* __restrict__ k,
                                                const _Float16* __restrict__ vT,
                                                _Float16* __restrict__ o) {
    const int lane  = threadIdx.x;
    const int qbase = blockIdx.x * 16;
    const int hh    = blockIdx.y;
    const int b     = blockIdx.z;

    __shared__ __align__(16) _Float16 Pt[16][32];   // exp(scores) tile
    __shared__ __align__(16) _Float16 Ks[32][64];   // K chunk (key-major)
    __shared__ float rsum_s[16];

    const int m = lane & 15;
    const _Float16* qrow = q + ((size_t)(b * S_ + qbase + m) * H_ + hh) * DK_;
    const v16h qf0 = frag_a(qrow, lane);        // d = 0..31
    const v16h qf1 = frag_a(qrow + 32, lane);   // d = 32..63

    const _Float16* vtb = vT + ((size_t)(b * H_ + hh) * 64) * S_;

    const v8f vzero = {0.f, 0.f, 0.f, 0.f, 0.f, 0.f, 0.f, 0.f};
    v8f oacc[4];
#pragma unroll
    for (int nt = 0; nt < 4; ++nt) oacc[nt] = vzero;
    float rs = 0.f;   // per-row softmax denominator (lanes 0-15 meaningful)

    const float scale = 0.125f;          // 1/sqrt(64)
    const int nkt = (qbase + 46) >> 5;   // tiles containing any k <= qbase+14

    for (int kt = 0; kt < nkt; ++kt) {
        const int kb = kt * 32;

        // Async-stage K chunk (32 keys x 64 d): lane copies its key row.
        {
            const _Float16* krow =
                k + ((size_t)(b * S_ + kb + lane) * H_ + hh) * DK_;
#pragma unroll
            for (int c8 = 0; c8 < 8; ++c8)
                async_copy_b128(&Ks[lane][c8 * 8], krow + c8 * 8);
            wait_asynccnt_0();   // single-wave block: wave-local completion
        }

        // Scores for two 16-key column tiles; mask+exp into Pt (f16)
#pragma unroll
        for (int c = 0; c < 2; ++c) {
            const _Float16* kcol = &Ks[c * 16 + (lane & 15)][0];
            const v16h kf0 = frag_b(kcol, lane);        // d = 0..31
            const v16h kf1 = frag_b(kcol + 32, lane);   // d = 32..63
            v8f sacc = vzero;
            sacc = wmma_16x16x32(qf0, kf0, sacc);
            sacc = wmma_16x16x32(qf1, kf1, sacc);

            const int coll  = c * 16 + (lane & 15);
            const int kglob = kb + coll;
            const int rbse  = (lane & 16) ? 8 : 0;
#pragma unroll
            for (int r = 0; r < 8; ++r) {
                const int rowt  = r + rbse;
                const int qglob = qbase + rowt;
                const float e =
                    (kglob < qglob) ? __expf(sacc[r] * scale) : 0.f;
                Pt[rowt][coll] = (_Float16)e;
            }
        }
        __syncthreads();

        // Row sums (denominator) — lanes 0-15 each own one query row
        if (lane < 16) {
            float s = 0.f;
#pragma unroll
            for (int j = 0; j < 32; ++j) s += (float)Pt[lane][j];
            rs += s;
        }

        // P (16x32) @ V (32x64): B-fragments straight from V^T (d-major)
        const v16h pf = frag_a(&Pt[lane & 15][0], lane);
#pragma unroll
        for (int nt = 0; nt < 4; ++nt) {
            const v16h vf =
                frag_b(vtb + (size_t)(nt * 16 + (lane & 15)) * S_ + kb, lane);
            oacc[nt] = wmma_16x16x32(pf, vf, oacc[nt]);
        }
        __syncthreads();
    }

    if (lane < 16) rsum_s[lane] = rs;
    __syncthreads();

    const int col = lane & 15;
    const int rb  = (lane & 16) ? 8 : 0;
#pragma unroll
    for (int nt = 0; nt < 4; ++nt)
#pragma unroll
        for (int r = 0; r < 8; ++r) {
            const int rowt = rb + r;
            const float den = rsum_s[rowt] + 1e-9f;
            o[((size_t)(b * S_ + qbase + rowt) * H_ + hh) * DK_ + nt * 16 + col] =
                (_Float16)(oacc[nt][r] / den);
        }
}

// ---------------------------------------------------------------------------
// Host-side orchestration
// ---------------------------------------------------------------------------
extern "C" void kernel_launch(void* const* d_in, const int* in_sizes, int n_in,
                              void* d_out, int out_size, void* d_ws, size_t ws_size,
                              hipStream_t stream) {
    const float* x  = (const float*)d_in[0];
    const float* wq = (const float*)d_in[1];
    const float* wk = (const float*)d_in[2];
    const float* wv = (const float*)d_in[3];
    const float* wo = (const float*)d_in[4];
    const float* wg = (const float*)d_in[5];
    const float* wu = (const float*)d_in[6];
    const float* wd = (const float*)d_in[7];
    const float* g1 = (const float*)d_in[8];
    const float* g2 = (const float*)d_in[9];
    float* out = (float*)d_out;

    // Carve workspace (256B aligned slices)
    char* ws = (char*)d_ws;
    size_t off = 0;
    auto carve = [&](size_t bytes) -> void* {
        void* p = ws + off;
        off = (off + bytes + 255) & ~(size_t)255;
        return p;
    };
    _Float16* h16   = (_Float16*)carve((size_t)ROWS * D_ * 2);
    _Float16* wqT   = (_Float16*)carve((size_t)D_ * D_ * 2);
    _Float16* wkT   = (_Float16*)carve((size_t)D_ * D_ * 2);
    _Float16* wvT   = (_Float16*)carve((size_t)D_ * D_ * 2);
    _Float16* woT   = (_Float16*)carve((size_t)D_ * D_ * 2);
    _Float16* wgT   = (_Float16*)carve((size_t)D_ * DFF_ * 2);
    _Float16* wuT   = (_Float16*)carve((size_t)D_ * DFF_ * 2);
    _Float16* wdT   = (_Float16*)carve((size_t)DFF_ * D_ * 2);
    _Float16* q16   = (_Float16*)carve((size_t)ROWS * D_ * 2);
    _Float16* k16   = (_Float16*)carve((size_t)ROWS * D_ * 2);
    _Float16* vT16  = (_Float16*)carve((size_t)ROWS * D_ * 2);  // [b][h][d][s]
    _Float16* av16  = (_Float16*)carve((size_t)ROWS * D_ * 2);
    float*    x1    = (float*)   carve((size_t)ROWS * D_ * 4);
    _Float16* h2    = (_Float16*)carve((size_t)ROWS * D_ * 2);
    _Float16* up16  = (_Float16*)carve((size_t)ROWS * DFF_ * 2); // reused: hidden

    // 1) Convert + transpose weights to f16 N-major [N][K]
    const int nD2 = D_ * D_;      // 1M
    const int nDF = D_ * DFF_;    // 4M
    cvt_transpose_f16<<<(nD2 + 255) / 256, 256, 0, stream>>>(wq, wqT, D_, D_);
    cvt_transpose_f16<<<(nD2 + 255) / 256, 256, 0, stream>>>(wk, wkT, D_, D_);
    cvt_transpose_f16<<<(nD2 + 255) / 256, 256, 0, stream>>>(wv, wvT, D_, D_);
    cvt_transpose_f16<<<(nD2 + 255) / 256, 256, 0, stream>>>(wo, woT, D_, D_);
    cvt_transpose_f16<<<(nDF + 255) / 256, 256, 0, stream>>>(wg, wgT, D_, DFF_);
    cvt_transpose_f16<<<(nDF + 255) / 256, 256, 0, stream>>>(wu, wuT, D_, DFF_);
    cvt_transpose_f16<<<(nDF + 255) / 256, 256, 0, stream>>>(wd, wdT, DFF_, D_);

    // 2) h = rmsnorm(x, g1)
    rmsnorm_f16<<<ROWS, 256, 0, stream>>>(x, g1, h16);

    // 3) Q/K/V projections (V written pre-transposed per head)
    dim3 gD(D_ / 128, ROWS / 128);      // (8, 32)
    gemm_wmma<0><<<gD, 256, 0, stream>>>(h16, wqT, q16, nullptr, nullptr, nullptr,
                                         ROWS, D_, D_);
    gemm_wmma<0><<<gD, 256, 0, stream>>>(h16, wkT, k16, nullptr, nullptr, nullptr,
                                         ROWS, D_, D_);
    gemm_wmma<3><<<gD, 256, 0, stream>>>(h16, wvT, vT16, nullptr, nullptr, nullptr,
                                         ROWS, D_, D_);

    // 4) Attention
    attn_wmma<<<dim3(S_ / 16, H_, B_), 32, 0, stream>>>(q16, k16, vT16, av16);

    // 5) x1 = x + av @ w_o
    gemm_wmma<1><<<gD, 256, 0, stream>>>(av16, woT, nullptr, x1, x, nullptr,
                                         ROWS, D_, D_);

    // 6) h2 = rmsnorm(x1, g2)
    rmsnorm_f16<<<ROWS, 256, 0, stream>>>(x1, g2, h2);

    // 7) FFN: up, then fused silu(gate)*up (in place over up16), then down+resid
    dim3 gF(DFF_ / 128, ROWS / 128);    // (32, 32)
    gemm_wmma<0><<<gF, 256, 0, stream>>>(h2, wuT, up16, nullptr, nullptr, nullptr,
                                         ROWS, DFF_, D_);
    gemm_wmma<2><<<gF, 256, 0, stream>>>(h2, wgT, up16, nullptr, nullptr, up16,
                                         ROWS, DFF_, D_);
    gemm_wmma<1><<<gD, 256, 0, stream>>>(up16, wdT, nullptr, out, x1, nullptr,
                                         ROWS, D_, DFF_);
}
